// multiHeadAttention_26422638805187
// MI455X (gfx1250) — matrix operand order repaired and checked by reading, not compiled, not run on hardware
//
#include <hip/hip_runtime.h>
#include <hip/hip_bf16.h>

typedef __bf16 bf16_t;
typedef __attribute__((ext_vector_type(16))) __bf16 v16bf;
typedef __attribute__((ext_vector_type(8)))  __bf16 v8bf;
typedef __attribute__((ext_vector_type(8)))  float  v8f;

union V8F { v8f v; float f[8]; };

constexpr int Bn = 4, Sn = 2048, En = 1024, Hn = 16, Dn = 64;
constexpr int ROWS = Bn * Sn;            // 8192 flattened (b,s) rows
constexpr int HALF = Dn / 2;             // 32 rope pairs

// ---------------------------------------------------------------- WMMA helpers
__device__ __forceinline__ v8f wmma_bf16(v16bf a, v16bf b, v8f c) {
  // (neg_a, A, neg_b, B, c_mod, C, reuse_a, reuse_b)
  return __builtin_amdgcn_wmma_f32_16x16x32_bf16(false, a, false, b, (short)0, c,
                                                 false, false);
}

// A-matrix fragment (16x32, bf16, row-major source, ld elements).
// lane l: m = l&15 ; k = 8*(l>=16) + e   (e<8)
//                    k = 8*(l>=16) + 8 + e (e>=8)  -> two contiguous 8-elt chunks
__device__ __forceinline__ v16bf load_a_frag(const bf16_t* base, int ld,
                                             int row0, int k0, int lane) {
  int m  = lane & 15;
  int kb = k0 + ((lane >> 4) << 3);
  const bf16_t* p = base + (size_t)(row0 + m) * ld + kb;
  union { v16bf v; v8bf h[2]; } u;
  u.h[0] = *(const v8bf*)(p);
  u.h[1] = *(const v8bf*)(p + 16);
  return u.v;
}

// B-matrix fragment (32x16, bf16). Source laid out so that row n holds the
// K-contiguous data of B column n (i.e. for C = A * Bsrc^T).
// lane l: n = n0 + (l&15) ; k = k0 + 8*(l>=16) + e (e<8), k0 + 8*(l>=16) + 8 + e (e>=8) -> two 8-elt chunks
__device__ __forceinline__ v16bf load_b_frag(const bf16_t* base, int ld,
                                             int n0, int k0, int lane) {
  int n  = n0 + (lane & 15);
  int kb = k0 + ((lane >> 4) << 3);
  const bf16_t* p = base + (size_t)n * ld + kb;
  union { v16bf v; v8bf h[2]; } u;
  u.h[0] = *(const v8bf*)(p);
  u.h[1] = *(const v8bf*)(p + 16);
  return u.v;
}

// ---------------------------------------------------------------- fp32 -> bf16
__global__ void cvt_bf16_kernel(const float* __restrict__ src,
                                bf16_t* __restrict__ dst, int n) {
  int i = blockIdx.x * blockDim.x + threadIdx.x;
  if (i < n) dst[i] = (bf16_t)src[i];
}

// ---------------------------------------------------------------- rope tables
__global__ void rope_table_kernel(float* __restrict__ cosT,
                                  float* __restrict__ sinT) {
  int i = blockIdx.x * blockDim.x + threadIdx.x;   // over S * HALF
  if (i >= Sn * HALF) return;
  int s = i / HALF, j = i % HALF;
  float inv = powf(10000.0f, -(float)j / (float)HALF);
  float a = (float)s * inv;
  cosT[i] = cosf(a);
  sinT[i] = sinf(a);
}

// ---------------------------------------------------------------- QKV + RoPE
// grid: (ROWS/128, H, 3). One wave -> 16 rows x 64 cols (one head slice).
// z=0: Q (+rope) -> [b,h,s,d] ; z=1: K (+rope) -> [b,h,s,d] ; z=2: V -> [b,h,d,s]
__global__ __launch_bounds__(256)
void qkv_kernel(const bf16_t* __restrict__ xb,
                const bf16_t* __restrict__ wq,
                const bf16_t* __restrict__ wk,
                const bf16_t* __restrict__ wv,
                const float*  __restrict__ cosT,
                const float*  __restrict__ sinT,
                bf16_t* __restrict__ qbuf,
                bf16_t* __restrict__ kbuf,
                bf16_t* __restrict__ vtbuf) {
  const int lane = threadIdx.x & 31;
  const int wave = threadIdx.x >> 5;
  const int row0 = blockIdx.x * 128 + wave * 16;
  const int h    = blockIdx.y;
  const int mat  = blockIdx.z;
  const bf16_t* W = (mat == 0) ? wq : (mat == 1) ? wk : wv;
  const int n0 = h * Dn;

  V8F acc[4];
  #pragma unroll
  for (int j = 0; j < 4; ++j)
    #pragma unroll
    for (int e = 0; e < 8; ++e) acc[j].f[e] = 0.0f;

  for (int k0 = 0; k0 < En; k0 += 32) {
    v16bf a = load_a_frag(xb, En, row0, k0, lane);
    #pragma unroll
    for (int j = 0; j < 4; ++j) {
      v16bf b = load_b_frag(W, En, n0 + j * 16, k0, lane);
      acc[j].v = wmma_bf16(a, b, acc[j].v);
    }
  }

  const int hi = lane >> 4, nn = lane & 15;
  if (mat < 2) {
    // RoPE: acc[j] holds d = j*16+nn ; pair (d, d+32) = (acc[j], acc[j+2])
    #pragma unroll
    for (int r = 0; r < 8; ++r) {
      int g = row0 + r + hi * 8;
      int s = g & (Sn - 1);
      #pragma unroll
      for (int j = 0; j < 2; ++j) {
        int d1 = j * 16 + nn;
        float c  = cosT[s * HALF + d1];
        float si = sinT[s * HALF + d1];
        float t1 = acc[j].f[r], t2 = acc[j + 2].f[r];
        acc[j].f[r]     = t1 * c - t2 * si;
        acc[j + 2].f[r] = t1 * si + t2 * c;
      }
    }
    bf16_t* dst = (mat == 0) ? qbuf : kbuf;
    #pragma unroll
    for (int r = 0; r < 8; ++r) {
      int g = row0 + r + hi * 8;
      int b = g >> 11, s = g & (Sn - 1);
      size_t base = (((size_t)b * Hn + h) * Sn + s) * Dn;
      #pragma unroll
      for (int j = 0; j < 4; ++j)
        dst[base + j * 16 + nn] = (bf16_t)acc[j].f[r];
    }
  } else {
    // V stored transposed: [b,h,d,s] so P@V B-fragments are K-contiguous
    #pragma unroll
    for (int r = 0; r < 8; ++r) {
      int g = row0 + r + hi * 8;
      int b = g >> 11, s = g & (Sn - 1);
      #pragma unroll
      for (int j = 0; j < 4; ++j) {
        int d = j * 16 + nn;
        vtbuf[(((size_t)b * Hn + h) * Dn + d) * Sn + s] = (bf16_t)acc[j].f[r];
      }
    }
  }
}

// ---------------------------------------------------------------- flash attention
// grid: (S/128, B*H). One wave owns 16 q rows; causal online softmax.
__global__ __launch_bounds__(256)
void attn_kernel(const bf16_t* __restrict__ qbuf,
                 const bf16_t* __restrict__ kbuf,
                 const bf16_t* __restrict__ vtbuf,
                 bf16_t* __restrict__ obuf) {
  __shared__ __align__(16) bf16_t pshm[8][16 * 32];  // per-wave P scratch
  const int lane = threadIdx.x & 31;
  const int wave = threadIdx.x >> 5;
  const int bh = blockIdx.y;
  const int b = bh >> 4, h = bh & 15;
  const int q0 = blockIdx.x * 128 + wave * 16;
  const bf16_t* Qh = qbuf  + (((size_t)b * Hn + h) * Sn) * Dn;
  const bf16_t* Kh = kbuf  + (((size_t)b * Hn + h) * Sn) * Dn;
  const bf16_t* Vt = vtbuf + (((size_t)b * Hn + h) * Dn) * Sn;
  const int hi = lane >> 4, nn = lane & 15;
  const float scale = 0.125f;  // D^-0.5, D=64

  // Q tile lives in registers for the whole kv loop (A-fragments, K-dim = 64)
  v16bf aq0 = load_a_frag(Qh, Dn, q0, 0,  lane);
  v16bf aq1 = load_a_frag(Qh, Dn, q0, 32, lane);

  V8F oacc[4];
  float m_i[8], l_i[8];
  #pragma unroll
  for (int r = 0; r < 8; ++r) { m_i[r] = -1e30f; l_i[r] = 0.0f; }
  #pragma unroll
  for (int j = 0; j < 4; ++j)
    #pragma unroll
    for (int e = 0; e < 8; ++e) oacc[j].f[e] = 0.0f;

  for (int kb = 0; kb <= q0 + 15; kb += 32) {
    // S = Q K^T  (two 16-col subtiles, K-dim 64 -> 2 wmma each)
    V8F s0, s1;
    #pragma unroll
    for (int e = 0; e < 8; ++e) { s0.f[e] = 0.0f; s1.f[e] = 0.0f; }
    {
      v16bf bk;
      bk = load_b_frag(Kh, Dn, kb,      0,  lane); s0.v = wmma_bf16(aq0, bk, s0.v);
      bk = load_b_frag(Kh, Dn, kb,      32, lane); s0.v = wmma_bf16(aq1, bk, s0.v);
      bk = load_b_frag(Kh, Dn, kb + 16, 0,  lane); s1.v = wmma_bf16(aq0, bk, s1.v);
      bk = load_b_frag(Kh, Dn, kb + 16, 32, lane); s1.v = wmma_bf16(aq1, bk, s1.v);
    }
    const bool need_mask = (kb + 31) > q0;

    #pragma unroll
    for (int r = 0; r < 8; ++r) {
      int row = q0 + r + hi * 8;
      float v0 = s0.f[r] * scale, v1 = s1.f[r] * scale;
      if (need_mask) {
        if (kb + nn > row)      v0 = -1e30f;
        if (kb + 16 + nn > row) v1 = -1e30f;
      }
      // row max across the 16 lanes holding this row (xor masks stay in-half)
      float mx = fmaxf(v0, v1);
      mx = fmaxf(mx, __shfl_xor(mx, 1, 32));
      mx = fmaxf(mx, __shfl_xor(mx, 2, 32));
      mx = fmaxf(mx, __shfl_xor(mx, 4, 32));
      mx = fmaxf(mx, __shfl_xor(mx, 8, 32));
      float m_new = fmaxf(m_i[r], mx);
      float alpha = __expf(m_i[r] - m_new);
      m_i[r] = m_new;
      float p0 = __expf(v0 - m_new);
      float p1 = __expf(v1 - m_new);
      float rs = p0 + p1;
      rs += __shfl_xor(rs, 1, 32);
      rs += __shfl_xor(rs, 2, 32);
      rs += __shfl_xor(rs, 4, 32);
      rs += __shfl_xor(rs, 8, 32);
      l_i[r] = l_i[r] * alpha + rs;
      #pragma unroll
      for (int j = 0; j < 4; ++j) oacc[j].f[r] *= alpha;
      // spill P (C layout) into LDS, to be re-read in A-fragment layout
      int prow = r + hi * 8;
      pshm[wave][prow * 32 + nn]      = (bf16_t)p0;
      pshm[wave][prow * 32 + 16 + nn] = (bf16_t)p1;
    }
    asm volatile("s_wait_dscnt 0" ::: "memory");  // same-wave LDS RAW

    // O += P @ V   (A = P 16x32, B = Vt rows are d-columns, K-contiguous)
    v16bf ap = load_a_frag(&pshm[wave][0], 32, 0, 0, lane);
    #pragma unroll
    for (int j = 0; j < 4; ++j) {
      v16bf bv = load_b_frag(Vt, Sn, j * 16, kb, lane);
      oacc[j].v = wmma_bf16(ap, bv, oacc[j].v);
    }
  }

  // normalize and emit O as bf16 rows of [B*S, E] (A-matrix of final GEMM)
  #pragma unroll
  for (int r = 0; r < 8; ++r) {
    float inv = 1.0f / l_i[r];
    int row = q0 + r + hi * 8;
    size_t base = ((size_t)b * Sn + row) * En + (size_t)h * Dn;
    #pragma unroll
    for (int j = 0; j < 4; ++j)
      obuf[base + j * 16 + nn] = (bf16_t)(oacc[j].f[r] * inv);
  }
}

// ---------------------------------------------------------------- out = O @ Wp^T + bp
__global__ __launch_bounds__(256)
void proj_kernel(const bf16_t* __restrict__ ob,
                 const bf16_t* __restrict__ wp,
                 const float*  __restrict__ bias,
                 float* __restrict__ out) {
  const int lane = threadIdx.x & 31;
  const int wave = threadIdx.x >> 5;
  const int row0 = blockIdx.x * 128 + wave * 16;
  const int n0   = blockIdx.y * 64;

  V8F acc[4];
  #pragma unroll
  for (int j = 0; j < 4; ++j)
    #pragma unroll
    for (int e = 0; e < 8; ++e) acc[j].f[e] = 0.0f;

  for (int k0 = 0; k0 < En; k0 += 32) {
    v16bf a = load_a_frag(ob, En, row0, k0, lane);
    #pragma unroll
    for (int j = 0; j < 4; ++j) {
      v16bf b = load_b_frag(wp, En, n0 + j * 16, k0, lane);
      acc[j].v = wmma_bf16(a, b, acc[j].v);
    }
  }
  const int hi = lane >> 4, nn = lane & 15;
  #pragma unroll
  for (int r = 0; r < 8; ++r) {
    int g = row0 + r + hi * 8;
    #pragma unroll
    for (int j = 0; j < 4; ++j) {
      int n = n0 + j * 16 + nn;
      out[(size_t)g * En + n] = acc[j].f[r] + bias[n];
    }
  }
}

// ---------------------------------------------------------------- launcher
extern "C" void kernel_launch(void* const* d_in, const int* in_sizes, int n_in,
                              void* d_out, int out_size, void* d_ws, size_t ws_size,
                              hipStream_t stream) {
  (void)in_sizes; (void)n_in; (void)out_size; (void)ws_size;
  const float* x  = (const float*)d_in[0];
  const float* Wq = (const float*)d_in[1];
  const float* Wk = (const float*)d_in[2];
  const float* Wv = (const float*)d_in[3];
  const float* Wp = (const float*)d_in[4];
  const float* bp = (const float*)d_in[5];
  float* out = (float*)d_out;

  char* p = (char*)d_ws;
  auto alloc = [&](size_t bytes) {
    char* r = p;
    p += (bytes + 255) & ~(size_t)255;
    return r;
  };
  bf16_t* xb   = (bf16_t*)alloc((size_t)ROWS * En * 2);
  bf16_t* wqb  = (bf16_t*)alloc((size_t)En * En * 2);
  bf16_t* wkb  = (bf16_t*)alloc((size_t)En * En * 2);
  bf16_t* wvb  = (bf16_t*)alloc((size_t)En * En * 2);
  bf16_t* wpb  = (bf16_t*)alloc((size_t)En * En * 2);
  bf16_t* qb   = (bf16_t*)alloc((size_t)Bn * Hn * Sn * Dn * 2);
  bf16_t* kb   = (bf16_t*)alloc((size_t)Bn * Hn * Sn * Dn * 2);
  bf16_t* vtb  = (bf16_t*)alloc((size_t)Bn * Hn * Sn * Dn * 2);
  bf16_t* obuf = (bf16_t*)alloc((size_t)ROWS * En * 2);
  float*  cosT = (float*)alloc((size_t)Sn * HALF * 4);
  float*  sinT = (float*)alloc((size_t)Sn * HALF * 4);

  int n = ROWS * En;
  cvt_bf16_kernel<<<(n + 255) / 256, 256, 0, stream>>>(x, xb, n);
  n = En * En;
  cvt_bf16_kernel<<<(n + 255) / 256, 256, 0, stream>>>(Wq, wqb, n);
  cvt_bf16_kernel<<<(n + 255) / 256, 256, 0, stream>>>(Wk, wkb, n);
  cvt_bf16_kernel<<<(n + 255) / 256, 256, 0, stream>>>(Wv, wvb, n);
  cvt_bf16_kernel<<<(n + 255) / 256, 256, 0, stream>>>(Wp, wpb, n);
  n = Sn * HALF;
  rope_table_kernel<<<(n + 255) / 256, 256, 0, stream>>>(cosT, sinT);

  qkv_kernel<<<dim3(ROWS / 128, Hn, 3), 256, 0, stream>>>(
      xb, wqb, wkb, wvb, cosT, sinT, qb, kb, vtb);
  attn_kernel<<<dim3(Sn / 128, Bn * Hn), 256, 0, stream>>>(qb, kb, vtb, obuf);
  proj_kernel<<<dim3(ROWS / 128, En / 64), 256, 0, stream>>>(obuf, wpb, bp, out);
}
